// Model_5549097746589
// MI455X (gfx1250) — matrix operand order repaired and checked by reading, not compiled, not run on hardware
//
#include <hip/hip_runtime.h>
#include <hip/hip_bf16.h>

typedef __attribute__((ext_vector_type(16))) _Float16 v16h;
typedef __attribute__((ext_vector_type(8)))  float    v8f;

#define Tlen 512
#define Bsz  512

__device__ inline float fast_sigmoid(float x) {
    return __builtin_amdgcn_rcpf(1.0f + __builtin_amdgcn_exp2f(-1.4426950408889634f * x));
}
__device__ inline float fast_tanh(float x) {
    return 2.0f * fast_sigmoid(2.0f * x) - 1.0f;
}

__device__ inline v8f wmma_f16(v16h a, v16h b, v8f c) {
    return __builtin_amdgcn_wmma_f32_16x16x32_f16(false, a, false, b, (short)0, c, false, false);
}

// Build A fragment (16xK f16, ktile kt) from LDS staging of h (row-major 16 x Kdim halves).
__device__ inline v16h load_a(const _Float16* hs, int Kdim, int kt, int lane) {
    int row = lane & 15;
    int kb  = (lane < 16) ? 0 : 8;
    const _Float16* p = hs + row * Kdim + kt * 32 + kb;
    v16h a;
#pragma unroll
    for (int j = 0; j < 8; ++j) { a[j] = p[j]; a[8 + j] = p[16 + j]; }
    return a;
}

// Load pre-swizzled B fragment: frag-major, lane-major, 16 contiguous halves.
__device__ inline v16h load_b(const _Float16* wf, int frag, int lane) {
    return *((const v16h*)(wf + ((size_t)frag * 32 + lane) * 16));
}

__device__ inline v8f bias_c(const float* bias, int nt, int lane) {
    float bv = bias[nt * 16 + (lane & 15)];
    v8f c;
#pragma unroll
    for (int i = 0; i < 8; ++i) c[i] = bv;
    return c;
}

// GRU gate update on one 16x16 C-layout tile triple.
__device__ inline v8f gru_tile(v8f xz, v8f xr, v8f xh, v8f hz, v8f hr, v8f hh, v8f hprev) {
    v8f out;
#pragma unroll
    for (int i = 0; i < 8; ++i) {
        float z = fast_sigmoid(xz[i] + hz[i]);
        float r = fast_sigmoid(xr[i] + hr[i]);
        float hc = fast_tanh(xh[i] + r * hh[i]);
        out[i] = z * hprev[i] + (1.0f - z) * hc;
    }
    return out;
}

// ---------------- Kernel 1: embedding + xp0 = inp @ k0 + bi0, written pre-swizzled ----
__global__ void embed_xp0_kernel(const float* __restrict__ x, const float* __restrict__ emb,
                                 const float* __restrict__ k0, const float* __restrict__ bi0,
                                 float* __restrict__ xp0s) {
    int t = blockIdx.x, b = blockIdx.y;
    int tid = threadIdx.x; // 192
    __shared__ float inp[19];
    const float* xr = x + ((size_t)b * Tlen + t) * 16;
    if (tid == 0) {
        inp[0] = xr[0];
    } else if (tid < 15) {
        inp[tid] = xr[tid + 1];           // rest = x[..., 2..15]
    } else if (tid < 19) {
        int id = (int)xr[1];
        inp[tid] = emb[id * 4 + (tid - 15)];
    }
    __syncthreads();
    float acc = bi0[tid];
#pragma unroll
    for (int k = 0; k < 19; ++k) acc += inp[k] * k0[k * 192 + tid];
    // scatter into WMMA C-layout swizzled store: [t][wave][ntile][lane][i]
    int w = b >> 4, m = b & 15;
    int nt = tid >> 4, col = tid & 15;
    int lane = col + ((m >= 8) ? 16 : 0);
    int i = m & 7;
    xp0s[((((size_t)t * 32 + w) * 12 + nt) * 32 + lane) * 8 + i] = acc;
}

// ---------------- Kernel 2: persistent recurrent GRU stack (1 wave per 16 batch rows) --
// frag directory (frag index bases into wfr):
//   R0: 0..23 (kt*12+nt), K1: 24..35 (kt*6+nt), R1: 36..41, K2: 42..47,
//   R2: 48..53, K3: 54..59, R3: 60..65
__global__ void __launch_bounds__(32, 1)
gru_scan_kernel(const float* __restrict__ xp0s,
                const float* __restrict__ r0, const float* __restrict__ k1,
                const float* __restrict__ r1, const float* __restrict__ k2,
                const float* __restrict__ r2, const float* __restrict__ k3,
                const float* __restrict__ r3,
                const float* __restrict__ bh0, const float* __restrict__ bi1,
                const float* __restrict__ bh1, const float* __restrict__ bi2,
                const float* __restrict__ bh2, const float* __restrict__ bi3,
                const float* __restrict__ bh3,
                float* __restrict__ outseq, float* __restrict__ hTbuf) {
    __shared__ _Float16 wfr[66 * 512];        // 67.5 KB of pre-swizzled B fragments
    __shared__ float bh0s[192];
    __shared__ float bi1s[96], bh1s[96], bi2s[96], bh2s[96], bi3s[96], bh3s[96];
    __shared__ _Float16 h0s[16 * 64], h1s[16 * 32], h2s[16 * 32], h3s[16 * 32];

    const int lane = threadIdx.x;   // block = 32 = one wave
    const int w = blockIdx.x;       // batch tile
    const int rbase = w * 16;

    // ---- init: swizzle weights f32->f16 fragments into LDS ----
    {
        const float* Wsrc[7] = {r0, k1, r1, k2, r2, k3, r3};
        const int Wn[7]   = {192, 96, 96, 96, 96, 96, 96};
        const int Wkt[7]  = {2, 2, 1, 1, 1, 1, 1};
        const int base[7] = {0, 24, 36, 42, 48, 54, 60};
        for (int mi = 0; mi < 7; ++mi) {
            int NT = Wn[mi] >> 4;
            int nf = NT * Wkt[mi];
            for (int f = 0; f < nf; ++f) {
                int kt = f / NT, nt = f % NT;
                for (int idx = lane; idx < 512; idx += 32) {
                    int li = idx >> 4, j = idx & 15;
                    int col = nt * 16 + (li & 15);
                    int krow = kt * 32 + ((li < 16) ? 0 : 8) + ((j < 8) ? j : (8 + j));
                    wfr[(size_t)(base[mi] + f) * 512 + li * 16 + j] =
                        (_Float16)Wsrc[mi][krow * Wn[mi] + col];
                }
            }
        }
        for (int i = lane; i < 192; i += 32) bh0s[i] = bh0[i];
        for (int i = lane; i < 96; i += 32) {
            bi1s[i] = bi1[i]; bh1s[i] = bh1[i];
            bi2s[i] = bi2[i]; bh2s[i] = bh2[i];
            bi3s[i] = bi3[i]; bh3s[i] = bh3[i];
        }
        for (int i = lane; i < 16 * 64; i += 32) h0s[i] = (_Float16)0.0f;
        for (int i = lane; i < 16 * 32; i += 32) {
            h1s[i] = (_Float16)0.0f; h2s[i] = (_Float16)0.0f; h3s[i] = (_Float16)0.0f;
        }
    }
    __syncthreads();

    v8f h0c[4], h1c[2], h2c[2], h3c[2];
#pragma unroll
    for (int g = 0; g < 4; ++g) h0c[g] = (v8f)0.0f;
#pragma unroll
    for (int g = 0; g < 2; ++g) { h1c[g] = (v8f)0.0f; h2c[g] = (v8f)0.0f; h3c[g] = (v8f)0.0f; }

    for (int t = 0; t < Tlen; ++t) {
        const float* xbase = xp0s + (((size_t)t * 32 + w) * 12) * 256 + (size_t)lane * 8;
        // prefetch xp0 tiles of the next batch tile (w+1) at this step (global_prefetch_b8)
        if (t + 1 < Tlen) {
#pragma unroll
            for (int nt = 0; nt < 12; ++nt)
                __builtin_prefetch(xbase + (size_t)(12 * 32 * 8) + (size_t)nt * 256, 0, 0);
        }

        // ================= layer 0: h0' = GRU(h0, xp0_t), per gate-group ==========
        {
            v16h a0 = load_a(h0s, 64, 0, lane);
            v16h a1 = load_a(h0s, 64, 1, lane);
#pragma unroll
            for (int g = 0; g < 4; ++g) {
                v8f hz = bias_c(bh0s, g, lane);
                hz = wmma_f16(a0, load_b(wfr, g, lane), hz);
                hz = wmma_f16(a1, load_b(wfr, 12 + g, lane), hz);
                v8f hr = bias_c(bh0s, 4 + g, lane);
                hr = wmma_f16(a0, load_b(wfr, 4 + g, lane), hr);
                hr = wmma_f16(a1, load_b(wfr, 16 + g, lane), hr);
                v8f hh = bias_c(bh0s, 8 + g, lane);
                hh = wmma_f16(a0, load_b(wfr, 8 + g, lane), hh);
                hh = wmma_f16(a1, load_b(wfr, 20 + g, lane), hh);
                v8f xz = *((const v8f*)(xbase + (size_t)g * 256));
                v8f xr = *((const v8f*)(xbase + (size_t)(4 + g) * 256));
                v8f xh = *((const v8f*)(xbase + (size_t)(8 + g) * 256));
                h0c[g] = gru_tile(xz, xr, xh, hz, hr, hh, h0c[g]);
            }
        }
        __syncthreads();
#pragma unroll
        for (int g = 0; g < 4; ++g)
#pragma unroll
            for (int i = 0; i < 8; ++i) {
                int m = (lane < 16) ? i : (8 + i);
                h0s[m * 64 + g * 16 + (lane & 15)] = (_Float16)h0c[g][i];
            }
        __syncthreads();

        // ================= layer 1, per gate-group =================
        {
            v16h ax0 = load_a(h0s, 64, 0, lane);
            v16h ax1 = load_a(h0s, 64, 1, lane);
            v16h ah  = load_a(h1s, 32, 0, lane);
#pragma unroll
            for (int g = 0; g < 2; ++g) {
                v8f xz = bias_c(bi1s, g, lane);
                xz = wmma_f16(ax0, load_b(wfr, 24 + g, lane), xz);
                xz = wmma_f16(ax1, load_b(wfr, 30 + g, lane), xz);
                v8f xr = bias_c(bi1s, 2 + g, lane);
                xr = wmma_f16(ax0, load_b(wfr, 26 + g, lane), xr);
                xr = wmma_f16(ax1, load_b(wfr, 32 + g, lane), xr);
                v8f xh = bias_c(bi1s, 4 + g, lane);
                xh = wmma_f16(ax0, load_b(wfr, 28 + g, lane), xh);
                xh = wmma_f16(ax1, load_b(wfr, 34 + g, lane), xh);
                v8f hz = bias_c(bh1s, g, lane);
                hz = wmma_f16(ah, load_b(wfr, 36 + g, lane), hz);
                v8f hr = bias_c(bh1s, 2 + g, lane);
                hr = wmma_f16(ah, load_b(wfr, 38 + g, lane), hr);
                v8f hh = bias_c(bh1s, 4 + g, lane);
                hh = wmma_f16(ah, load_b(wfr, 40 + g, lane), hh);
                h1c[g] = gru_tile(xz, xr, xh, hz, hr, hh, h1c[g]);
            }
        }
        __syncthreads();
#pragma unroll
        for (int g = 0; g < 2; ++g)
#pragma unroll
            for (int i = 0; i < 8; ++i) {
                int m = (lane < 16) ? i : (8 + i);
                h1s[m * 32 + g * 16 + (lane & 15)] = (_Float16)h1c[g][i];
            }
        __syncthreads();

        // ================= layer 2, per gate-group =================
        {
            v16h ax = load_a(h1s, 32, 0, lane);
            v16h ah = load_a(h2s, 32, 0, lane);
#pragma unroll
            for (int g = 0; g < 2; ++g) {
                v8f xz = bias_c(bi2s, g, lane);
                xz = wmma_f16(ax, load_b(wfr, 42 + g, lane), xz);
                v8f xr = bias_c(bi2s, 2 + g, lane);
                xr = wmma_f16(ax, load_b(wfr, 44 + g, lane), xr);
                v8f xh = bias_c(bi2s, 4 + g, lane);
                xh = wmma_f16(ax, load_b(wfr, 46 + g, lane), xh);
                v8f hz = bias_c(bh2s, g, lane);
                hz = wmma_f16(ah, load_b(wfr, 48 + g, lane), hz);
                v8f hr = bias_c(bh2s, 2 + g, lane);
                hr = wmma_f16(ah, load_b(wfr, 50 + g, lane), hr);
                v8f hh = bias_c(bh2s, 4 + g, lane);
                hh = wmma_f16(ah, load_b(wfr, 52 + g, lane), hh);
                h2c[g] = gru_tile(xz, xr, xh, hz, hr, hh, h2c[g]);
            }
        }
        __syncthreads();
#pragma unroll
        for (int g = 0; g < 2; ++g)
#pragma unroll
            for (int i = 0; i < 8; ++i) {
                int m = (lane < 16) ? i : (8 + i);
                h2s[m * 32 + g * 16 + (lane & 15)] = (_Float16)h2c[g][i];
            }
        __syncthreads();

        // ================= layer 3, per gate-group =================
        {
            v16h ax = load_a(h2s, 32, 0, lane);
            v16h ah = load_a(h3s, 32, 0, lane);
#pragma unroll
            for (int g = 0; g < 2; ++g) {
                v8f xz = bias_c(bi3s, g, lane);
                xz = wmma_f16(ax, load_b(wfr, 54 + g, lane), xz);
                v8f xr = bias_c(bi3s, 2 + g, lane);
                xr = wmma_f16(ax, load_b(wfr, 56 + g, lane), xr);
                v8f xh = bias_c(bi3s, 4 + g, lane);
                xh = wmma_f16(ax, load_b(wfr, 58 + g, lane), xh);
                v8f hz = bias_c(bh3s, g, lane);
                hz = wmma_f16(ah, load_b(wfr, 60 + g, lane), hz);
                v8f hr = bias_c(bh3s, 2 + g, lane);
                hr = wmma_f16(ah, load_b(wfr, 62 + g, lane), hr);
                v8f hh = bias_c(bh3s, 4 + g, lane);
                hh = wmma_f16(ah, load_b(wfr, 64 + g, lane), hh);
                h3c[g] = gru_tile(xz, xr, xh, hz, hr, hh, h3c[g]);
            }
        }
        __syncthreads();
#pragma unroll
        for (int g = 0; g < 2; ++g)
#pragma unroll
            for (int i = 0; i < 8; ++i) {
                int m = (lane < 16) ? i : (8 + i);
                h3s[m * 32 + g * 16 + (lane & 15)] = (_Float16)h3c[g][i];
            }
        __syncthreads();

        // emit h3 for this step
#pragma unroll
        for (int g = 0; g < 2; ++g)
#pragma unroll
            for (int i = 0; i < 8; ++i) {
                int m = (lane < 16) ? i : (8 + i);
                outseq[((size_t)(rbase + m) * Tlen + t) * 32 + g * 16 + (lane & 15)] = h3c[g][i];
            }
    }

    // final hidden state hT = h3(T-1)
#pragma unroll
    for (int g = 0; g < 2; ++g)
#pragma unroll
        for (int i = 0; i < 8; ++i) {
            int m = (lane < 16) ? i : (8 + i);
            hTbuf[(size_t)(rbase + m) * 32 + g * 16 + (lane & 15)] = h3c[g][i];
        }
}

// ---------------- Kernel 3: attention pooling + output head -----------------
__global__ void attn_head_kernel(const float* __restrict__ outseq, const float* __restrict__ hT,
                                 const float* __restrict__ w1, const float* __restrict__ b1,
                                 const float* __restrict__ w2, const float* __restrict__ b2,
                                 const float* __restrict__ v, const float* __restrict__ bv,
                                 const float* __restrict__ wd, const float* __restrict__ bd,
                                 float* __restrict__ out) {
    int b = blockIdx.x;
    int tid = threadIdx.x; // 256
    __shared__ float w1s[1024], b1s[32], vs[32], hterm[32], svals[Tlen], red[256], ctxs[32];
    for (int i = tid; i < 1024; i += 256) w1s[i] = w1[i];
    if (tid < 32) { b1s[tid] = b1[tid]; vs[tid] = v[tid]; }
    __syncthreads();
    if (tid < 32) {
        float acc = b2[tid];
        for (int k = 0; k < 32; ++k) acc += hT[(size_t)b * 32 + k] * w2[k * 32 + tid];
        hterm[tid] = acc;
    }
    __syncthreads();
    for (int t = tid; t < Tlen; t += 256) {
        const float* o = outseq + ((size_t)b * Tlen + t) * 32;
        float s = bv[0];
        for (int n = 0; n < 32; ++n) {
            float acc = b1s[n] + hterm[n];
            for (int k = 0; k < 32; ++k) acc += o[k] * w1s[k * 32 + n];
            s += fast_tanh(acc) * vs[n];
        }
        svals[t] = s;
    }
    __syncthreads();
    float m = -1e30f;
    for (int t = tid; t < Tlen; t += 256) m = fmaxf(m, svals[t]);
    red[tid] = m; __syncthreads();
    for (int s2 = 128; s2 > 0; s2 >>= 1) {
        if (tid < s2) red[tid] = fmaxf(red[tid], red[tid + s2]);
        __syncthreads();
    }
    float mx = red[0];
    __syncthreads();
    float psum = 0.0f;
    for (int t = tid; t < Tlen; t += 256) {
        float e = __builtin_amdgcn_exp2f((svals[t] - mx) * 1.4426950408889634f);
        svals[t] = e; psum += e;
    }
    red[tid] = psum; __syncthreads();
    for (int s2 = 128; s2 > 0; s2 >>= 1) {
        if (tid < s2) red[tid] += red[tid + s2];
        __syncthreads();
    }
    float rsum = __builtin_amdgcn_rcpf(red[0]);
    __syncthreads();
    if (tid < 32) {
        float acc = 0.0f;
        for (int t = 0; t < Tlen; ++t) acc += svals[t] * outseq[((size_t)b * Tlen + t) * 32 + tid];
        ctxs[tid] = acc * rsum;
    }
    __syncthreads();
    if (tid == 0) {
        float d0 = bd[0], d1 = bd[1];
        for (int n = 0; n < 32; ++n) { d0 += ctxs[n] * wd[n * 2]; d1 += ctxs[n] * wd[n * 2 + 1]; }
        float mm = fmaxf(d0, d1);
        float e0 = __builtin_amdgcn_exp2f((d0 - mm) * 1.4426950408889634f);
        float e1 = __builtin_amdgcn_exp2f((d1 - mm) * 1.4426950408889634f);
        float inv = __builtin_amdgcn_rcpf(e0 + e1);
        out[b * 2] = e0 * inv;
        out[b * 2 + 1] = e1 * inv;
    }
}

extern "C" void kernel_launch(void* const* d_in, const int* in_sizes, int n_in,
                              void* d_out, int out_size, void* d_ws, size_t ws_size,
                              hipStream_t stream) {
    const float* x   = (const float*)d_in[0];
    const float* emb = (const float*)d_in[1];
    const float* k0  = (const float*)d_in[2];
    const float* r0  = (const float*)d_in[3];
    const float* bi0 = (const float*)d_in[4];
    const float* bh0 = (const float*)d_in[5];
    const float* k1  = (const float*)d_in[6];
    const float* r1  = (const float*)d_in[7];
    const float* bi1 = (const float*)d_in[8];
    const float* bh1 = (const float*)d_in[9];
    const float* k2  = (const float*)d_in[10];
    const float* r2  = (const float*)d_in[11];
    const float* bi2 = (const float*)d_in[12];
    const float* bh2 = (const float*)d_in[13];
    const float* k3  = (const float*)d_in[14];
    const float* r3  = (const float*)d_in[15];
    const float* bi3 = (const float*)d_in[16];
    const float* bh3 = (const float*)d_in[17];
    const float* w1  = (const float*)d_in[18];
    const float* b1  = (const float*)d_in[19];
    const float* w2  = (const float*)d_in[20];
    const float* b2  = (const float*)d_in[21];
    const float* v   = (const float*)d_in[22];
    const float* bv  = (const float*)d_in[23];
    const float* wd  = (const float*)d_in[24];
    const float* bd  = (const float*)d_in[25];

    float* xp0s   = (float*)d_ws;                               // B*T*192 f32 (swizzled)
    float* outseq = xp0s + (size_t)Bsz * Tlen * 192;            // B*T*32 f32
    float* hTbuf  = outseq + (size_t)Bsz * Tlen * 32;           // B*32 f32

    dim3 g1(Tlen, Bsz);
    embed_xp0_kernel<<<g1, 192, 0, stream>>>(x, emb, k0, bi0, xp0s);
    gru_scan_kernel<<<Bsz / 16, 32, 0, stream>>>(xp0s, r0, k1, r1, k2, r2, k3, r3,
                                                 bh0, bi1, bh1, bi2, bh2, bi3, bh3,
                                                 outseq, hTbuf);
    attn_head_kernel<<<Bsz, 256, 0, stream>>>(outseq, hTbuf, w1, b1, w2, b2, v, bv, wd, bd,
                                              (float*)d_out);
}
